// Self_Attn_31370441130462
// MI455X (gfx1250) — matrix operand order repaired and checked by reading, not compiled, not run on hardware
//
#include <hip/hip_runtime.h>
#include <hip/hip_bf16.h>

// ---------------------------------------------------------------------------
// Self-attention for MI455X (gfx1250, wave32, WMMA).
//   q = x@Wq^T+bq ; k = x@Wk^T+bk ; v = x@Wv^T+bv
//   out = softmax(q k^T) v        (no 1/sqrt(d) scaling, matches reference)
// ~95 GFLOP vs ~100MB min HBM traffic -> matrix-unit bound -> bf16 WMMA with
// f32 accumulation + f32 softmax. All bf16 intermediates (~85MB) fit the
// 192MB L2. Pipeline:
//   0) elementwise f32->bf16 convert of x and Wq/Wk/Wv (one pass, native cvt)
//   1) WMMA QKV projection (bf16 fragments, 3-deep software-pipelined loads)
//   2) flash attention (S via WMMA, online softmax f32, PV via WMMA)
// B-fragment streams are triple-buffered: two 2-load clauses are in flight
// when each WMMA issues, so waits are partial instead of full stalls.
// ---------------------------------------------------------------------------

#define B_DIM   8
#define S_DIM   2048
#define D_DIM   512

typedef __attribute__((ext_vector_type(16))) __bf16 v16bf;
typedef __attribute__((ext_vector_type(8)))  float  v8f;

// Fragment: 8 VGPRs = 16 bf16 per lane.
union ABfrag {
    v16bf        v;
    unsigned int u[8];
};

__device__ __forceinline__ unsigned short f2bf(float f) {
    __bf16 b = (__bf16)f;                       // native cvt op
    return __builtin_bit_cast(unsigned short, b);
}

__device__ __forceinline__ unsigned int pack2bf(float a, float b) {
    return (unsigned int)f2bf(a) | ((unsigned int)f2bf(b) << 16);
}

// Load 8 consecutive bf16 (16B) each from p0 and p1 into the fragment.
// Works for global or LDS pointers (addrspace inferred after inlining).
__device__ __forceinline__ void load2x16B(ABfrag& f, const unsigned short* p0,
                                          const unsigned short* p1) {
    uint4 a = *reinterpret_cast<const uint4*>(p0);
    uint4 b = *reinterpret_cast<const uint4*>(p1);
    f.u[0] = a.x; f.u[1] = a.y; f.u[2] = a.z; f.u[3] = a.w;
    f.u[4] = b.x; f.u[5] = b.y; f.u[6] = b.z; f.u[7] = b.w;
}

__device__ __forceinline__ v8f wmma_bf16(const ABfrag& a, const ABfrag& b, v8f c) {
    // D = A(16x32) * B(32x16) + C(16x16 f32)
    return __builtin_amdgcn_wmma_f32_16x16x32_bf16(
        false, a.v, false, b.v, (short)0, c, false, false);
}

// ---------------------------------------------------------------------------
// Kernel 0: elementwise f32 -> bf16, 8 elements (one uint4 out) per thread.
// ---------------------------------------------------------------------------
__global__ __launch_bounds__(256) void cvt_bf16_kernel(
    const float* __restrict__ src, unsigned short* __restrict__ dst, int n8) {
    const int i = blockIdx.x * 256 + threadIdx.x;
    if (i >= n8) return;
    const float4* s = reinterpret_cast<const float4*>(src) + 2 * (size_t)i;
    const float4 a = s[0];
    const float4 b = s[1];
    uint4 o;
    o.x = pack2bf(a.x, a.y);
    o.y = pack2bf(a.z, a.w);
    o.z = pack2bf(b.x, b.y);
    o.w = pack2bf(b.z, b.w);
    reinterpret_cast<uint4*>(dst)[i] = o;
}

// ---------------------------------------------------------------------------
// Kernel 1: fused QKV projection (all operands already bf16).
//   grid = (rows/128, cols/128, 3 matrices), block = 256 (8 waves).
//   Wave w computes a 16x128 output stripe (8 WMMA n-tiles), K-loop over 512.
//   A-frag: xb rows; B-frag: Wb rows (W is [out,in], so B[k][n] = W[n][k] is
//   16 contiguous bf16 per lane).  A double-buffered across k-chunks,
//   B triple-buffered across n-tiles (2 clauses in flight per WMMA).
//   Q,K stored row-major bf16; V stored transposed [b][d][s] bf16 so the
//   attention PV B-fragments are contiguous.
// ---------------------------------------------------------------------------
__global__ __launch_bounds__(256) void qkv_proj_kernel(
    const unsigned short* __restrict__ xb, const unsigned short* __restrict__ Wb,
    const float* __restrict__ bq, const float* __restrict__ bk,
    const float* __restrict__ bv,
    unsigned short* __restrict__ qo, unsigned short* __restrict__ ko,
    unsigned short* __restrict__ vTo) {
    const int mat = blockIdx.z;
    const unsigned short* W = Wb + (size_t)mat * D_DIM * D_DIM;
    const float* bias = (mat == 0) ? bq : (mat == 1) ? bk : bv;

    const int w    = threadIdx.x >> 5;
    const int lane = threadIdx.x & 31;
    const int hi   = lane >> 4;        // lane half selects K sub-runs
    const int lo16 = lane & 15;

    const int rowBase = blockIdx.x * 128 + w * 16;  // 16 rows per wave
    const int colBase = blockIdx.y * 128;           // 128 cols per block

    v8f acc[8] = {};  // 8 n-tiles of 16x16 f32

    const unsigned short* xrow  = xb + (size_t)(rowBase + lo16) * D_DIM;
    // B base for n-tile 0; n-tile nt adds nt*16 rows of W.
    const unsigned short* wbase = W + (size_t)(colBase + lo16) * D_DIM + hi * 16;
    const int aoff = hi ? 16 : 0;

    ABfrag a[2], bfr[3];
    load2x16B(a[0], xrow + aoff, xrow + aoff + 8);  // A(kc=0)

    for (int kc = 0; kc < D_DIM; kc += 32) {
        const int idx = (kc >> 5) & 1;
        if (kc + 32 < D_DIM) {  // preload next A; overlaps the 8 WMMAs below
            const int off1 = kc + 32 + aoff;
            load2x16B(a[idx ^ 1], xrow + off1, xrow + off1 + 8);
        }
        // prime two B fragments (nt=0,1)
        load2x16B(bfr[0], wbase + kc, wbase + kc + 8);
        {
            const unsigned short* p = wbase + (size_t)16 * D_DIM + kc;
            load2x16B(bfr[1], p, p + 8);
        }
#pragma unroll
        for (int nt = 0; nt < 8; ++nt) {
            if (nt < 6) {  // keep 2 fragment clauses in flight
                const unsigned short* p =
                    wbase + (size_t)(nt + 2) * 16 * D_DIM + kc;
                load2x16B(bfr[(nt + 2) % 3], p, p + 8);
            }
            acc[nt] = wmma_bf16(a[idx], bfr[nt % 3], acc[nt]);
        }
    }

    // bias add + bf16 store (D layout: M = i + 8*hi, N = lo16)
#pragma unroll
    for (int nt = 0; nt < 8; ++nt) {
        const int col = colBase + nt * 16 + lo16;
        const float bval = bias[col];
#pragma unroll
        for (int i = 0; i < 8; ++i) {
            const int r = rowBase + i + 8 * hi;   // flat row in [0, 16384)
            const unsigned short bf = f2bf(acc[nt][i] + bval);
            if (mat == 2) {
                const int bb = r >> 11, s = r & (S_DIM - 1);
                vTo[(size_t)bb * D_DIM * S_DIM + (size_t)col * S_DIM + s] = bf;
            } else if (mat == 0) {
                qo[(size_t)r * D_DIM + col] = bf;
            } else {
                ko[(size_t)r * D_DIM + col] = bf;
            }
        }
    }
}

// ---------------------------------------------------------------------------
// Kernel 2: flash attention.
//   grid = (2048/32 q-tiles, 8 batches), block = 256 (8 waves).
//   Per 128-key tile:
//     S phase : wave w computes S[32 x 16keys] (keys 16w..16w+15) via WMMA,
//               fragments loaded straight from global bf16 Q/K -> LDS f32.
//               Whole fragment set double-buffered across k-chunks.
//     softmax : 8 threads/row, shfl_xor reductions, online max/sum,
//               P = exp(S - m_new) stored bf16 in LDS.
//     PV phase: wave w owns d-cols [64w,64w+64): O = O*alpha + P * V,
//               A-frags from LDS (ds_load_b128), B-frags from V^T global,
//               B triple-buffered across n-tiles.
//   Final: out = O / l  (f32).
// ---------------------------------------------------------------------------
__global__ __launch_bounds__(256) void flash_attn_kernel(
    const unsigned short* __restrict__ q, const unsigned short* __restrict__ k,
    const unsigned short* __restrict__ vT, float* __restrict__ out) {
    __shared__ __align__(16) float          Sld[32 * 128];
    __shared__ __align__(16) unsigned short Pld[32 * 128];
    __shared__ float mrow[32], lrow[32], arow[32];

    const int b     = blockIdx.y;
    const int qBase = blockIdx.x * 32;           // query rows within batch
    const int w     = threadIdx.x >> 5;
    const int lane  = threadIdx.x & 31;
    const int hi    = lane >> 4;
    const int lo16  = lane & 15;

    const unsigned short* qb = q  + (size_t)b * S_DIM * D_DIM;
    const unsigned short* kb = k  + (size_t)b * S_DIM * D_DIM;
    const unsigned short* vb = vT + (size_t)b * D_DIM * S_DIM;

    if (threadIdx.x < 32) {
        mrow[threadIdx.x] = -__builtin_inff();
        lrow[threadIdx.x] = 0.0f;
        arow[threadIdx.x] = 1.0f;
    }
    __syncthreads();

    v8f o[2][4] = {};  // [m-tile][n-tile]: 32 rows x 64 d-cols per wave

    const unsigned short* qr0 = qb + (size_t)(qBase + lo16) * D_DIM;
    const unsigned short* qr1 = qb + (size_t)(qBase + 16 + lo16) * D_DIM;
    const int aoff = hi ? 16 : 0;

    for (int kt = 0; kt < S_DIM; kt += 128) {
        // ---- S = Q K^T (this wave: 16 keys) ----
        v8f s0 = {}, s1 = {};
        const unsigned short* kr =
            kb + (size_t)(kt + w * 16 + lo16) * D_DIM + hi * 16;
        __builtin_prefetch(kb + (size_t)(kt + 128 + w * 16 + lo16) * D_DIM, 0, 0);

        ABfrag a0[2], a1[2], bf[2];
        // prologue: fragment set for kc=0
        load2x16B(a0[0], qr0 + aoff, qr0 + aoff + 8);
        load2x16B(a1[0], qr1 + aoff, qr1 + aoff + 8);
        load2x16B(bf[0], kr, kr + 8);

        for (int kc = 0; kc < D_DIM; kc += 32) {
            const int idx = (kc >> 5) & 1;
            if (kc + 32 < D_DIM) {  // preload next set; overlaps 2 WMMAs
                const int off1 = kc + 32 + aoff;
                load2x16B(a0[idx ^ 1], qr0 + off1, qr0 + off1 + 8);
                load2x16B(a1[idx ^ 1], qr1 + off1, qr1 + off1 + 8);
                load2x16B(bf[idx ^ 1], kr + kc + 32, kr + kc + 40);
            }
            s0 = wmma_bf16(a0[idx], bf[idx], s0);
            s1 = wmma_bf16(a1[idx], bf[idx], s1);
        }
#pragma unroll
        for (int i = 0; i < 8; ++i) {
            Sld[(i + 8 * hi) * 128 + w * 16 + lo16]      = s0[i];
            Sld[(16 + i + 8 * hi) * 128 + w * 16 + lo16] = s1[i];
        }
        __syncthreads();

        // ---- online softmax over this 128-key tile ----
        {
            const int r = threadIdx.x >> 3;   // 32 rows, 8 threads per row
            const int sub = threadIdx.x & 7;  // 16 cols each
            float sv[16];
            float tmax = -__builtin_inff();
#pragma unroll
            for (int c = 0; c < 16; ++c) {
                sv[c] = Sld[r * 128 + sub * 16 + c];
                tmax = fmaxf(tmax, sv[c]);
            }
            tmax = fmaxf(tmax, __shfl_xor(tmax, 1, 32));
            tmax = fmaxf(tmax, __shfl_xor(tmax, 2, 32));
            tmax = fmaxf(tmax, __shfl_xor(tmax, 4, 32));
            const float mprev = mrow[r];
            const float mnew  = fmaxf(mprev, tmax);
            const float alpha = __expf(mprev - mnew);
            float psum = 0.0f;
#pragma unroll
            for (int c = 0; c < 16; ++c) {
                const float e = __expf(sv[c] - mnew);
                psum += e;
                Pld[r * 128 + sub * 16 + c] = f2bf(e);
            }
            psum += __shfl_xor(psum, 1, 32);
            psum += __shfl_xor(psum, 2, 32);
            psum += __shfl_xor(psum, 4, 32);
            if (sub == 0) {
                lrow[r] = lrow[r] * alpha + psum;
                mrow[r] = mnew;
                arow[r] = alpha;
            }
        }
        __syncthreads();

        // ---- rescale O by alpha, then O += P * V ----
#pragma unroll
        for (int mt = 0; mt < 2; ++mt) {
#pragma unroll
            for (int i = 0; i < 8; ++i) {
                const float al = arow[mt * 16 + i + 8 * hi];
#pragma unroll
                for (int nt = 0; nt < 4; ++nt) o[mt][nt][i] *= al;
            }
        }
        const unsigned short* vrb = vb + (size_t)(w * 64 + lo16) * S_DIM + kt;
        for (int ks = 0; ks < 128; ks += 32) {
            ABfrag pa0, pa1, vf[3];
            const int off1 = ks + aoff;
            load2x16B(pa0, &Pld[lo16 * 128 + off1], &Pld[lo16 * 128 + off1 + 8]);
            load2x16B(pa1, &Pld[(16 + lo16) * 128 + off1],
                      &Pld[(16 + lo16) * 128 + off1 + 8]);
            // prime two V fragments (nt=0,1)
            load2x16B(vf[0], vrb + ks + hi * 16, vrb + ks + hi * 16 + 8);
            {
                const unsigned short* p =
                    vrb + (size_t)16 * S_DIM + ks + hi * 16;
                load2x16B(vf[1], p, p + 8);
            }
#pragma unroll
            for (int nt = 0; nt < 4; ++nt) {
                if (nt < 2) {  // keep 2 fragment clauses in flight
                    const unsigned short* vr =
                        vrb + (size_t)(nt + 2) * 16 * S_DIM + ks + hi * 16;
                    load2x16B(vf[(nt + 2) % 3], vr, vr + 8);
                }
                o[0][nt] = wmma_bf16(pa0, vf[nt % 3], o[0][nt]);
                o[1][nt] = wmma_bf16(pa1, vf[nt % 3], o[1][nt]);
            }
        }
        __syncthreads();
    }

    // ---- normalize and store f32 output ----
#pragma unroll
    for (int mt = 0; mt < 2; ++mt) {
#pragma unroll
        for (int i = 0; i < 8; ++i) {
            const int row = mt * 16 + i + 8 * hi;
            const float invl = 1.0f / lrow[row];
#pragma unroll
            for (int nt = 0; nt < 4; ++nt) {
                const int dcol = w * 64 + nt * 16 + lo16;
                out[((size_t)b * S_DIM + qBase + row) * D_DIM + dcol] =
                    o[mt][nt][i] * invl;
            }
        }
    }
}

// ---------------------------------------------------------------------------
extern "C" void kernel_launch(void* const* d_in, const int* in_sizes, int n_in,
                              void* d_out, int out_size, void* d_ws,
                              size_t ws_size, hipStream_t stream) {
    const float* x  = (const float*)d_in[0];
    const float* Wq = (const float*)d_in[1];
    const float* bq = (const float*)d_in[2];
    const float* Wk = (const float*)d_in[3];
    const float* bk = (const float*)d_in[4];
    const float* Wv = (const float*)d_in[5];
    const float* bv = (const float*)d_in[6];
    float* out = (float*)d_out;

    const size_t NX = (size_t)B_DIM * S_DIM * D_DIM;   // 8.39M elements
    const size_t NW = (size_t)D_DIM * D_DIM;           // 262144 elements

    // workspace layout (bf16 = unsigned short):
    //   xb[NX] | Wb[3*NW] | Q[NX] | K[NX] | V^T[NX]
    unsigned short* xb = (unsigned short*)d_ws;
    unsigned short* Wb = xb + NX;
    unsigned short* qw = Wb + 3 * NW;
    unsigned short* kw = qw + NX;
    unsigned short* vw = kw + NX;

    // 0) one-time f32 -> bf16 conversions
    cvt_bf16_kernel<<<(int)(NX / 8 / 256), 256, 0, stream>>>(x, xb, (int)(NX / 8));
    cvt_bf16_kernel<<<(int)(NW / 8 / 256), 256, 0, stream>>>(Wq, Wb, (int)(NW / 8));
    cvt_bf16_kernel<<<(int)(NW / 8 / 256), 256, 0, stream>>>(Wk, Wb + NW, (int)(NW / 8));
    cvt_bf16_kernel<<<(int)(NW / 8 / 256), 256, 0, stream>>>(Wv, Wb + 2 * NW, (int)(NW / 8));

    // 1) QKV projection
    dim3 g1((B_DIM * S_DIM) / 128, D_DIM / 128, 3);
    qkv_proj_kernel<<<g1, 256, 0, stream>>>(xb, Wb, bq, bk, bv, qw, kw, vw);

    // 2) flash attention
    dim3 g2(S_DIM / 32, B_DIM);
    flash_attn_kernel<<<g2, 256, 0, stream>>>(qw, kw, vw, out);
}
